// BasicLSTM_59021440582009
// MI455X (gfx1250) — matrix operand order repaired and checked by reading, not compiled, not run on hardware
//
#include <hip/hip_runtime.h>
#include <hip/hip_bf16.h>
#include <math.h>

#define SEQ    512
#define BATCH  64
#define HID    1024
#define LAYERS 2
#define KDIM   (2 * HID)   // 2048
#define NDIM   (4 * HID)   // 4096
#define NBLK   64          // persistent workgroups (one per 16-col hidden strip)

typedef __attribute__((ext_vector_type(16))) __bf16        v16bf;
typedef __attribute__((ext_vector_type(8)))  float         v8f;
typedef __attribute__((ext_vector_type(8)))  float         f8;
typedef __attribute__((ext_vector_type(8)))  unsigned int  v8u;

// ---------------------------------------------------------------------------
// Branch-free transcendentals on CDNA5 TRANS pipes (no EXEC-mask divergence
// on the recurrence critical path).
// ---------------------------------------------------------------------------
__device__ __forceinline__ float fast_sigmoid(float x) {
    // 1 / (1 + 2^(-x*log2(e)))  ->  v_exp_f32 + v_rcp_f32
    float e = __builtin_amdgcn_exp2f(-x * 1.4426950408889634f);
    return __builtin_amdgcn_rcpf(1.0f + e);
}

__device__ __forceinline__ float fast_tanh(float x) {
#if __has_builtin(__builtin_amdgcn_tanhf)
    return __builtin_amdgcn_tanhf(x);        // v_tanh_f32 (gfx1250 TRANS op)
#else
    return 2.0f * fast_sigmoid(2.0f * x) - 1.0f;
#endif
}

// ---------------------------------------------------------------------------
// Repack W (f32 row-major [K=2048, N=4096] per layer) into bf16 fragments laid
// out as B operands of v_wmma_f32_16x16x32_bf16, in the contiguous K order the A side uses:
//   fragment(ns, kc): 32 lanes x 16 bf16 (32 bytes contiguous per lane).
//   lane L: column n = ns*16 + (L&15); K run = kc*32 + (L<16 ? 0 : 16) .. +15
// Flat layout: Wb[(((l*256 + ns)*64 + kc)*32 + L)*16 + e]
// ---------------------------------------------------------------------------
__global__ __launch_bounds__(256) void pack_w(const float* __restrict__ W,
                                              __bf16* __restrict__ Wb) {
    int tid  = blockIdx.x * blockDim.x + threadIdx.x;   // < LAYERS*256*64*32
    int lane = tid & 31;
    int kc   = (tid >> 5) & 63;
    int ns   = (tid >> 11) & 255;
    int l    = tid >> 19;
    if (l >= LAYERS) return;

    const float* w = W + (size_t)l * KDIM * NDIM;
    __bf16*      o = Wb + (size_t)tid * 16;
    int n     = ns * 16 + (lane & 15);
    int kbase = kc * 32 + ((lane & 16) ? 16 : 0);
#pragma unroll
    for (int e = 0; e < 16; ++e) {
        o[e] = (__bf16)w[(size_t)(kbase + e) * NDIM + n];
    }
}

// ---------------------------------------------------------------------------
// Device-wide barrier for the persistent kernel: monotonic counter scheme
// (no reset races). Release-fence publishes this block's stores, acquire
// fence after the spin makes every lane's subsequent loads coherent.
// ---------------------------------------------------------------------------
__device__ __forceinline__ void grid_barrier(unsigned* cnt, unsigned target) {
    __builtin_amdgcn_fence(__ATOMIC_RELEASE, "agent");   // publish our stores
    __syncthreads();
    if (threadIdx.x == 0) {
        __hip_atomic_fetch_add(cnt, 1u, __ATOMIC_ACQ_REL,
                               __HIP_MEMORY_SCOPE_AGENT);
        while (__hip_atomic_load(cnt, __ATOMIC_ACQUIRE,
                                 __HIP_MEMORY_SCOPE_AGENT) < target) {
            __builtin_amdgcn_s_sleep(4);
        }
    }
    __syncthreads();
    __builtin_amdgcn_fence(__ATOMIC_ACQUIRE, "agent");   // invalidate stale
}

// ---------------------------------------------------------------------------
// One LSTM layer phase: gates = concat(x, h_prev) @ W + b, then cell update.
// Wave (wv) owns a 16-row M tile; block (wg) owns a 16-wide hidden column
// strip; 4 f32 accumulators (gates i/j/f/o at column offsets 0/H/2H/3H) so
// the nonlinearity fuses in-register with zero extra global traffic.
// ---------------------------------------------------------------------------
__device__ __forceinline__ void layer_phase(
        const float*  __restrict__ x,      // [64,1024] layer input
        const float*  __restrict__ hprev,  // [64,1024] previous hidden
        const __bf16* __restrict__ Wb,     // packed bf16 weights (layer base)
        const float*  __restrict__ bias,   // [4096]
        float* __restrict__ hout,          // [64,1024] new hidden
        float* __restrict__ c,             // [64,1024] cell state (in/out)
        float* __restrict__ yout,          // optional sequence output
        int lane, int wv, int wg)
{
    const int nsub = lane & 15;
    const int hi   = (lane >> 4);             // 0/1 lane half
    const int mrow = wv * 16 + nsub;          // A-matrix row for this lane
    const int arun = (lane & 16) ? 16 : 0;    // A K-run start within chunk

    v8f acc[4];
#pragma unroll
    for (int g = 0; g < 4; ++g)
#pragma unroll
        for (int e = 0; e < 8; ++e) acc[g][e] = 0.0f;

    // K loop: 2048 = 64 chunks of 32. First 1024 from x, second from hprev.
#pragma unroll 2
    for (int kc = 0; kc < KDIM / 32; ++kc) {
        const int    kk  = kc * 32;
        const float* src = (kk < HID) ? (x + (size_t)mrow * HID)
                                      : (hprev + (size_t)mrow * HID);
        const int    kb  = (kk & (HID - 1)) + arun;

        // A fragment: lane holds K run [kb..kb+15]
        f8 a0 = *(const f8*)(src + kb);
        f8 a1 = *(const f8*)(src + kb + 8);
        v16bf a;
#pragma unroll
        for (int e = 0; e < 8; ++e) a[e] = (__bf16)a0[e];
#pragma unroll
        for (int e = 0; e < 8; ++e) a[8 + e] = (__bf16)a1[e];

#pragma unroll
        for (int g = 0; g < 4; ++g) {
            const __bf16* bp =
                Wb + ((size_t)((g * 64 + wg) * 64 + kc) * 32 + lane) * 16;
            v8u   braw  = *(const v8u*)bp;           // 32B contiguous per lane
            v16bf bfrag = __builtin_bit_cast(v16bf, braw);
            acc[g] = __builtin_amdgcn_wmma_f32_16x16x32_bf16(
                false, a, false, bfrag, (short)0, acc[g], false, false);
        }
    }

    // Fused LSTM cell. C/D layout: VGPR r, lane L -> (M = r + 8*(L>>4),
    // N = L&15) within the 16x16 tile.
    const int   hcol = wg * 16 + nsub;
    const float bi = bias[0 * HID + hcol];
    const float bj = bias[1 * HID + hcol];
    const float bf = bias[2 * HID + hcol];
    const float bo = bias[3 * HID + hcol];

#pragma unroll
    for (int r = 0; r < 8; ++r) {
        const int    m   = wv * 16 + r + hi * 8;
        const size_t idx = (size_t)m * HID + hcol;
        float gi = acc[0][r] + bi;
        float gj = acc[1][r] + bj;
        float gf = acc[2][r] + bf;
        float go = acc[3][r] + bo;
        float si = fast_sigmoid(gi);
        float sf = fast_sigmoid(gf);
        float so = fast_sigmoid(go);
        float cp = c[idx];
        float cn = cp * sf + si * fast_tanh(gj);
        float hn = fast_tanh(cn) * so;
        c[idx]    = cn;
        hout[idx] = hn;
        if (yout) yout[idx] = hn;
    }
}

// ---------------------------------------------------------------------------
// Persistent kernel: runs the entire 512-step, 2-layer recurrence in ONE
// launch. Software grid barriers separate the dependent phases, eliminating
// ~1024 kernel dispatch/drain latencies from the critical path.
// ---------------------------------------------------------------------------
__global__ __launch_bounds__(128) void lstm_persistent(
        const float*  __restrict__ input,  // [512,64,1024]
        const float*  __restrict__ bias,   // [2,4096]
        const __bf16* __restrict__ Wb,     // packed bf16 weights, both layers
        float* __restrict__ hA,            // [2,64,1024] ping
        float* __restrict__ hB,            // [2,64,1024] pong
        float* __restrict__ cbuf,          // [2,64,1024] cell state
        float* __restrict__ yout,          // [512,64,1024]
        unsigned* __restrict__ bar)        // zeroed barrier counter
{
    const int lane = threadIdx.x & 31;
    const int wv   = threadIdx.x >> 5;     // 0..3 : M tile
    const int wg   = blockIdx.x;           // 0..63: hidden column strip

    const __bf16* Wb1 = Wb + (size_t)KDIM * NDIM;
    unsigned nbar = 0;

    for (int t = 0; t < SEQ; ++t) {
        const float* xt = input + (size_t)t * BATCH * HID;
        float* hR = (t & 1) ? hB : hA;     // read  h_{t-1}
        float* hW = (t & 1) ? hA : hB;     // write h_t

        // Layer 0
        layer_phase(xt, hR, Wb, bias, hW, cbuf, nullptr, lane, wv, wg);
        grid_barrier(bar, NBLK * (++nbar));

        // Layer 1 (input = layer 0's fresh hidden)
        layer_phase(hW, hR + BATCH * HID, Wb1, bias + NDIM,
                    hW + BATCH * HID, cbuf + BATCH * HID,
                    yout + (size_t)t * BATCH * HID, lane, wv, wg);
        grid_barrier(bar, NBLK * (++nbar));
    }
}

// ---------------------------------------------------------------------------
extern "C" void kernel_launch(void* const* d_in, const int* in_sizes, int n_in,
                              void* d_out, int out_size, void* d_ws,
                              size_t ws_size, hipStream_t stream) {
    const float* input = (const float*)d_in[0];  // [512,64,1024]
    const float* h0    = (const float*)d_in[1];  // [2,64,1024]
    const float* c0    = (const float*)d_in[2];  // [2,64,1024]
    const float* W     = (const float*)d_in[3];  // [2,2048,4096]
    const float* b     = (const float*)d_in[4];  // [2,4096]
    float*       out   = (float*)d_out;          // y[512,64,1024] ++ h ++ c

    const size_t stateElems = (size_t)LAYERS * BATCH * HID;   // 131072
    const size_t wbElems    = (size_t)LAYERS * KDIM * NDIM;   // 16,777,216

    char*     ws   = (char*)d_ws;
    __bf16*   Wb   = (__bf16*)ws;                             // 32 MB
    float*    hA   = (float*)(ws + wbElems * sizeof(__bf16));
    float*    hB   = hA + stateElems;
    float*    cbuf = hB + stateElems;
    unsigned* bar  = (unsigned*)(cbuf + stateElems);

    // Initialize recurrent state + barrier (graph-capture-safe async ops).
    hipMemcpyAsync(hA, h0, stateElems * sizeof(float),
                   hipMemcpyDeviceToDevice, stream);
    hipMemcpyAsync(cbuf, c0, stateElems * sizeof(float),
                   hipMemcpyDeviceToDevice, stream);
    hipMemsetAsync(bar, 0, sizeof(unsigned), stream);

    // Repack weights to bf16 WMMA fragments (L2-resident thereafter).
    const int packThreads = LAYERS * 256 * 64 * 32;  // 1,048,576
    pack_w<<<packThreads / 256, 256, 0, stream>>>(W, Wb);

    // Entire recurrence in one persistent launch.
    lstm_persistent<<<NBLK, 128, 0, stream>>>(input, b, Wb, hA, hB, cbuf,
                                              out, bar);

    // SEQ is even: final states live in hA. Emit (last_hidden, last_cell).
    hipMemcpyAsync(out + (size_t)SEQ * BATCH * HID, hA,
                   stateElems * sizeof(float), hipMemcpyDeviceToDevice, stream);
    hipMemcpyAsync(out + (size_t)SEQ * BATCH * HID + stateElems, cbuf,
                   stateElems * sizeof(float), hipMemcpyDeviceToDevice, stream);
}
